// PreNormedHGTCavAttention_43885975831006
// MI455X (gfx1250) — matrix operand order repaired and checked by reading, not compiled, not run on hardware
//
#include <hip/hip_runtime.h>
#include <hip/hip_bf16.h>
#include <stdint.h>

// Problem constants (match reference)
#define B_  2
#define L_  5
#define H_  48
#define W_  48
#define C_  256
#define M_  8
#define D_  32
#define NT_ 2
#define NR_ 4
#define HW_ (H_*W_)
#define PT_ (HW_/16)                 // 144 pixel tiles
#define SCALE_ 0.1767766952966369f   // 32^-0.5

typedef __attribute__((ext_vector_type(16))) __bf16 v16bf;
typedef __attribute__((ext_vector_type(8)))  __bf16 v8bf;
typedef __attribute__((ext_vector_type(8)))  float   v8f;
typedef __attribute__((__vector_size__(16))) int     i32x4;

#define GLOBAL_AS __attribute__((address_space(1)))
#define LDS_AS    __attribute__((address_space(3)))

#if defined(__AMDGCN__) && __has_builtin(__builtin_amdgcn_global_load_async_to_lds_b128)
#define HAVE_ASYNC_LDS 1
#else
#define HAVE_ASYNC_LDS 0
#endif

__device__ __forceinline__ unsigned short f2bf(float f) {
  unsigned int u = __float_as_uint(f);
  u += 0x7FFFu + ((u >> 16) & 1u);       // round-to-nearest-even
  return (unsigned short)(u >> 16);
}
__device__ __forceinline__ float bf2f(unsigned short h) {
  return __uint_as_float(((unsigned int)h) << 16);
}
// prior_encoding[b][l][0][0][2] -> node type
__device__ __forceinline__ int node_type(const float* pe, int b, int l) {
  return (int)pe[(size_t)(b*L_ + l) * (H_*W_*3) + 2];
}

// Stage one contiguous 8KB (16 rows x 256 bf16) tile into LDS; blockDim.x must be 512.
__device__ __forceinline__ void stage_tile_8k(unsigned short* lds, const unsigned short* g) {
#if HAVE_ASYNC_LDS
  {
    unsigned t = threadIdx.x;                       // 512 threads x 16B = 8192B
    GLOBAL_AS i32x4* gp = (GLOBAL_AS i32x4*)((GLOBAL_AS char*)g + t * 16);
    LDS_AS    i32x4* lp = (LDS_AS    i32x4*)((LDS_AS    char*)lds + t * 16);
    __builtin_amdgcn_global_load_async_to_lds_b128(gp, lp, 0, 0);
  }
#if __has_builtin(__builtin_amdgcn_s_wait_asynccnt)
  __builtin_amdgcn_s_wait_asynccnt(0);
#else
  asm volatile("s_wait_asynccnt 0" ::: "memory");
#endif
#else
  {
    unsigned t = threadIdx.x;
    ((uint4*)lds)[t] = ((const uint4*)g)[t];
  }
#endif
  __syncthreads();
}

// Build a v16bf A-fragment (16-bit 16x32 WMMA A layout) from a bf16 row pointer.
__device__ __forceinline__ v16bf load_afrag(const unsigned short* row) {
  v8bf a0 = *(const v8bf*)(row);
  v8bf a1 = *(const v8bf*)(row + 8);
  return __builtin_shufflevector(a0, a1, 0,1,2,3,4,5,6,7,8,9,10,11,12,13,14,15);
}

// ---------------- Kernel 1: weight conversion fp32 -> bf16 (+ both rel transposes) ---
__global__ void wconv_kernel(const float* __restrict__ qw, const float* __restrict__ kw,
                             const float* __restrict__ vw, const float* __restrict__ aw,
                             const float* __restrict__ ra, const float* __restrict__ rm,
                             unsigned short* __restrict__ qwb, unsigned short* __restrict__ kwb,
                             unsigned short* __restrict__ vwb, unsigned short* __restrict__ awb,
                             unsigned short* __restrict__ raT, unsigned short* __restrict__ rmT) {
  int i = blockIdx.x * blockDim.x + threadIdx.x;
  const int NW = NT_ * C_ * C_;                 // 131072
  if (i < NW) {
    qwb[i] = f2bf(qw[i]);
    kwb[i] = f2bf(kw[i]);
    vwb[i] = f2bf(vw[i]);
    awb[i] = f2bf(aw[i]);
  }
  const int NRW = NR_ * M_ * D_ * D_;           // 32768
  if (i < NRW) {
    int q = i % D_;
    int p = (i / D_) % D_;
    int rmm = i / (D_ * D_);
    size_t oT = ((size_t)rmm * D_ + q) * D_ + p;  // [rel][m][q][p]
    raT[oT] = f2bf(ra[i]);                        // transpose last two dims
    rmT[oT] = f2bf(rm[i]);
  }
}

// ---------------- Kernel 2: LayerNorm -> bf16, one wave per C=256 row ----------------
__global__ void ln_bf16_kernel(const float* __restrict__ x,
                               const float* __restrict__ gw,
                               const float* __restrict__ gb,
                               unsigned short* __restrict__ xn) {
  const int wpb  = blockDim.x >> 5;
  int wave = blockIdx.x * wpb + (threadIdx.x >> 5);
  int lane = threadIdx.x & 31;
  const int rows = B_ * L_ * HW_;
  if (wave >= rows) return;
  const float* xr = x + (size_t)wave * C_;
  float v[8]; float s = 0.f, ss = 0.f;
#pragma unroll
  for (int i = 0; i < 8; ++i) { float t = xr[lane + 32*i]; v[i] = t; s += t; ss += t*t; }
#pragma unroll
  for (int off = 16; off; off >>= 1) {          // wave32 tree reduction
    s  += __shfl_xor(s,  off, 32);
    ss += __shfl_xor(ss, off, 32);
  }
  float mu   = s * (1.f / C_);
  float rstd = rsqrtf(ss * (1.f / C_) - mu * mu + 1e-5f);
  unsigned short* orow = xn + (size_t)wave * C_;
#pragma unroll
  for (int i = 0; i < 8; ++i) {
    int c = lane + 32*i;
    orow[c] = f2bf((v[i] - mu) * rstd * gw[c] + gb[c]);
  }
}

// ---------------- Kernel 3: typed Q/K/V projection, WMMA 16x16x32 bf16 ---------------
// One 512-thread block per (tensor, b*l, pixel-tile): A tile staged once in LDS (async),
// 16 waves each own one 16-output column tile; K-loop over C=256 (8 WMMAs/wave).
__global__ void qkv_wmma_kernel(const unsigned short* __restrict__ xn,
                                const float* __restrict__ pe,
                                const unsigned short* __restrict__ qwb,
                                const unsigned short* __restrict__ kwb,
                                const unsigned short* __restrict__ vwb,
                                const float* __restrict__ qbias,
                                const float* __restrict__ kbias,
                                const float* __restrict__ vbias,
                                unsigned short* __restrict__ qo,
                                unsigned short* __restrict__ ko,
                                unsigned short* __restrict__ vo) {
  int blk = blockIdx.x;
  int pt = blk % PT_;        blk /= PT_;
  int bl = blk % (B_ * L_);  blk /= (B_ * L_);
  int which = blk;                                 // 0=q 1=k 2=v
  int nt   = threadIdx.x >> 5;                     // 16 column tiles
  int lane = threadIdx.x & 31;
  int b = bl / L_, l = bl % L_;
  int t = node_type(pe, b, l);

  __shared__ __align__(64) unsigned short atile[16 * C_];
  int p0 = pt * 16;
  stage_tile_8k(atile, xn + ((size_t)bl * HW_ + p0) * C_);

  const unsigned short* Wt  = (which==0 ? qwb : which==1 ? kwb : vwb) + (size_t)t * C_ * C_;
  const float*          bia = (which==0 ? qbias : which==1 ? kbias : vbias) + (size_t)t * C_;
  unsigned short*       out = (which==0 ? qo : which==1 ? ko : vo);

  int mr = lane & 15, hi = lane >> 4;
  int n0 = nt * 16;
  const unsigned short* arow = atile + mr * C_ + hi * 16;          // LDS (ds_load)
  const unsigned short* brow = Wt + (size_t)(n0 + mr) * C_ + hi * 16;

  v8f acc = {};
#pragma unroll
  for (int kk = 0; kk < C_; kk += 32) {
    __builtin_prefetch(brow + kk + 64, 0, 1);     // -> global_prefetch_b8
    v16bf A  = load_afrag(arow + kk);
    v16bf Bm = *(const v16bf*)(brow + kk);
    acc = __builtin_amdgcn_wmma_f32_16x16x32_bf16(false, A, false, Bm, (short)0, acc,
                                                  false, false);
  }
  int col = n0 + mr;
  float bv = bia[col];
#pragma unroll
  for (int r = 0; r < 8; ++r) {                   // C/D layout: VGPR r -> M = r + hi*8
    int row = p0 + r + hi * 8;
    out[((size_t)bl * HW_ + row) * C_ + col] = f2bf(acc[r] + bv);
  }
}

// ---------------- Kernel 4: FUSED relation transform + attention + combine -----------
// One wave per (b, head, 16-pixel tile). QW = q@W_att (WMMA, K=32 exact), att via
// row-wise dot against k using wave32 shfl_xor tree, softmax, msg = v@W_msg (WMMA),
// output accumulated in C/D layout. kp/msg intermediates never touch HBM.
__global__ void fused_attn_kernel(const unsigned short* __restrict__ qb,
                                  const unsigned short* __restrict__ kb,
                                  const unsigned short* __restrict__ vb,
                                  const unsigned short* __restrict__ raT, // [rel][m][q][p]
                                  const unsigned short* __restrict__ rmT, // [rel][m][c][p]
                                  const int* __restrict__ mask,
                                  const float* __restrict__ pe,
                                  unsigned short* __restrict__ oa) {
  const int wpb = blockDim.x >> 5;
  int wid  = blockIdx.x * wpb + (threadIdx.x >> 5);
  int lane = threadIdx.x & 31;
  int pt = wid % PT_; wid /= PT_;
  int m  = wid % M_;  wid /= M_;
  int b  = wid;
  if (b >= B_) return;

  int ty[L_];
#pragma unroll
  for (int l = 0; l < L_; ++l) ty[l] = node_type(pe, b, l);

  int mr = lane & 15, hi = lane >> 4;
  int p0 = pt * 16;

  // cache V A-fragments for all j (reused by every i)
  v16bf VA[L_];
#pragma unroll
  for (int j = 0; j < L_; ++j)
    VA[j] = load_afrag(vb + ((size_t)(b*L_ + j) * HW_ + p0 + mr) * C_ + m * D_ + hi * 16);

#pragma unroll 1
  for (int i = 0; i < L_; ++i) {
    v16bf QA = load_afrag(qb + ((size_t)(b*L_ + i) * HW_ + p0 + mr) * C_ + m * D_ + hi * 16);

    // QW for both possible t_j values (rel = t_i*NT + t_j), two 16-col halves each
    v8f qw[NT_][2];
#pragma unroll
    for (int tj = 0; tj < NT_; ++tj) {
      int rel = ty[i] * NT_ + tj;
      const unsigned short* WB = raT + (size_t)(rel * M_ + m) * D_ * D_;
#pragma unroll
      for (int half = 0; half < 2; ++half) {
        v16bf Bm = *(const v16bf*)(WB + (size_t)(half*16 + mr) * D_ + hi * 16);
        v8f z = {};
        qw[tj][half] = __builtin_amdgcn_wmma_f32_16x16x32_bf16(false, QA, false, Bm,
                                                               (short)0, z, false, false);
      }
    }

    // attention scores: row-wise dot(QW, k_j) across 32 cols = 2 halves x 16 lanes
    v8f att8[L_];
#pragma unroll
    for (int j = 0; j < L_; ++j) {
      int tj = ty[j];
      const unsigned short* krow = kb + ((size_t)(b*L_ + j) * HW_) * C_ + m * D_;
      v8f part = {};
#pragma unroll
      for (int r = 0; r < 8; ++r) {
        size_t pbase = (size_t)(p0 + r + hi*8) * C_;
        part[r] = qw[tj][0][r] * bf2f(krow[pbase + mr])
                + qw[tj][1][r] * bf2f(krow[pbase + 16 + mr]);
      }
#pragma unroll
      for (int off = 1; off < 16; off <<= 1) {     // reduce within 16-lane group
#pragma unroll
        for (int r = 0; r < 8; ++r) part[r] += __shfl_xor(part[r], off, 32);
      }
#pragma unroll
      for (int r = 0; r < 8; ++r) {
        int pix = p0 + r + hi*8;
        int mk = mask[((size_t)b * HW_ + pix) * (L_*L_) + i*L_ + j];
        part[r] = (mk == 0) ? -1e9f : part[r] * SCALE_;
      }
      att8[j] = part;
    }

    // softmax over j, per pixel row
    v8f mx = att8[0], den = {};
#pragma unroll
    for (int j = 1; j < L_; ++j)
#pragma unroll
      for (int r = 0; r < 8; ++r) mx[r] = fmaxf(mx[r], att8[j][r]);
#pragma unroll
    for (int j = 0; j < L_; ++j)
#pragma unroll
      for (int r = 0; r < 8; ++r) { att8[j][r] = __expf(att8[j][r] - mx[r]); den[r] += att8[j][r]; }
    v8f inv;
#pragma unroll
    for (int r = 0; r < 8; ++r) inv[r] = 1.f / den[r];

    // combine: out += att * (v_j @ W_msg), all in C/D layout
    v8f o0 = {}, o1 = {};
#pragma unroll
    for (int j = 0; j < L_; ++j) {
      int rel = ty[i] * NT_ + ty[j];
      const unsigned short* WM = rmT + (size_t)(rel * M_ + m) * D_ * D_;
      v16bf B0 = *(const v16bf*)(WM + (size_t)(mr) * D_ + hi * 16);
      v16bf B1 = *(const v16bf*)(WM + (size_t)(16 + mr) * D_ + hi * 16);
      v8f z0 = {}, z1 = {};
      v8f m0 = __builtin_amdgcn_wmma_f32_16x16x32_bf16(false, VA[j], false, B0,
                                                       (short)0, z0, false, false);
      v8f m1 = __builtin_amdgcn_wmma_f32_16x16x32_bf16(false, VA[j], false, B1,
                                                       (short)0, z1, false, false);
#pragma unroll
      for (int r = 0; r < 8; ++r) {
        float w = att8[j][r] * inv[r];
        o0[r] += w * m0[r];
        o1[r] += w * m1[r];
      }
    }
#pragma unroll
    for (int r = 0; r < 8; ++r) {
      size_t base = ((size_t)(b*L_ + i) * HW_ + p0 + r + hi*8) * C_ + m * D_;
      oa[base + mr]      = f2bf(o0[r]);
      oa[base + 16 + mr] = f2bf(o1[r]);
    }
  }
}

// ---------------- Kernel 5: typed output projection, WMMA, fp32 result ---------------
__global__ void outproj_wmma_kernel(const unsigned short* __restrict__ oa,
                                    const float* __restrict__ pe,
                                    const unsigned short* __restrict__ awb,
                                    const float* __restrict__ abias,
                                    float* __restrict__ out) {
  int blk = blockIdx.x;
  int pt = blk % PT_;  blk /= PT_;
  int bl = blk;                                    // (b*L + l)
  int nt   = threadIdx.x >> 5;
  int lane = threadIdx.x & 31;
  int b = bl / L_, l = bl % L_;
  int t = node_type(pe, b, l);

  __shared__ __align__(64) unsigned short atile[16 * C_];
  int p0 = pt * 16;
  stage_tile_8k(atile, oa + ((size_t)bl * HW_ + p0) * C_);

  const unsigned short* Wt = awb + (size_t)t * C_ * C_;   // a_weight[t][c][i], K contiguous
  int mr = lane & 15, hi = lane >> 4;
  int n0 = nt * 16;
  const unsigned short* arow = atile + mr * C_ + hi * 16;
  const unsigned short* brow = Wt + (size_t)(n0 + mr) * C_ + hi * 16;

  v8f acc = {};
#pragma unroll
  for (int kk = 0; kk < C_; kk += 32) {
    __builtin_prefetch(brow + kk + 64, 0, 1);
    v16bf A  = load_afrag(arow + kk);
    v16bf Bm = *(const v16bf*)(brow + kk);
    acc = __builtin_amdgcn_wmma_f32_16x16x32_bf16(false, A, false, Bm, (short)0, acc,
                                                  false, false);
  }
  int col = n0 + mr;
  float bv = abias[(size_t)t * C_ + col];
#pragma unroll
  for (int r = 0; r < 8; ++r) {
    int row = p0 + r + hi * 8;
    out[((size_t)bl * HW_ + row) * C_ + col] = acc[r] + bv;   // (B,L,H,W,C) layout
  }
}

// ------------------------------------ launcher ---------------------------------------
extern "C" void kernel_launch(void* const* d_in, const int* in_sizes, int n_in,
                              void* d_out, int out_size, void* d_ws, size_t ws_size,
                              hipStream_t stream) {
  (void)in_sizes; (void)n_in; (void)out_size; (void)ws_size;
  const float* x    = (const float*)d_in[0];
  const int*   mask = (const int*)  d_in[1];
  const float* pe   = (const float*)d_in[2];
  const float* lnw  = (const float*)d_in[3];
  const float* lnb  = (const float*)d_in[4];
  const float* qw   = (const float*)d_in[5];
  const float* qbi  = (const float*)d_in[6];
  const float* kw   = (const float*)d_in[7];
  const float* kbi  = (const float*)d_in[8];
  const float* vw   = (const float*)d_in[9];
  const float* vbi  = (const float*)d_in[10];
  const float* aw   = (const float*)d_in[11];
  const float* abi  = (const float*)d_in[12];
  const float* ra   = (const float*)d_in[13];
  const float* rm   = (const float*)d_in[14];
  float* out = (float*)d_out;

  char* ws = (char*)d_ws;
  size_t off = 0;
  auto alloc_bf = [&](size_t elems) -> unsigned short* {
    unsigned short* p = (unsigned short*)(ws + off);
    off += ((elems * 2) + 255) & ~(size_t)255;
    return p;
  };
  const size_t ACT = (size_t)B_ * L_ * HW_ * C_;            // 5,898,240
  unsigned short* xn   = alloc_bf(ACT);
  unsigned short* qb   = alloc_bf(ACT);
  unsigned short* kb   = alloc_bf(ACT);
  unsigned short* vb   = alloc_bf(ACT);
  unsigned short* oa   = alloc_bf(ACT);
  unsigned short* qwb  = alloc_bf((size_t)NT_ * C_ * C_);
  unsigned short* kwb  = alloc_bf((size_t)NT_ * C_ * C_);
  unsigned short* vwb  = alloc_bf((size_t)NT_ * C_ * C_);
  unsigned short* awb  = alloc_bf((size_t)NT_ * C_ * C_);
  unsigned short* raT  = alloc_bf((size_t)NR_ * M_ * D_ * D_);
  unsigned short* rmT  = alloc_bf((size_t)NR_ * M_ * D_ * D_);

  // 1. weight conversion (+ relation transposes)
  {
    int n = NT_ * C_ * C_;
    wconv_kernel<<<(n + 255) / 256, 256, 0, stream>>>(qw, kw, vw, aw, ra, rm,
                                                      qwb, kwb, vwb, awb, raT, rmT);
  }
  // 2. LayerNorm
  {
    int rows = B_ * L_ * HW_;                               // 23040
    ln_bf16_kernel<<<(rows + 3) / 4, 128, 0, stream>>>(x, lnw, lnb, xn);
  }
  // 3. Q/K/V projections: one block per (tensor, bl, ptile), 512 threads
  {
    int blocks = 3 * (B_ * L_) * PT_;                       // 4320
    qkv_wmma_kernel<<<blocks, 512, 0, stream>>>(
        xn, pe, qwb, kwb, vwb, qbi, kbi, vbi, qb, kb, vb);
  }
  // 4. fused relation + attention + combine: one wave per (b, m, ptile)
  {
    int waves = B_ * M_ * PT_;                              // 2304
    fused_attn_kernel<<<(waves + 3) / 4, 128, 0, stream>>>(
        qb, kb, vb, raT, rmT, mask, pe, oa);
  }
  // 5. output projection -> fp32 (B,L,H,W,C)
  {
    int blocks = (B_ * L_) * PT_;                           // 1440
    outproj_wmma_kernel<<<blocks, 512, 0, stream>>>(oa, pe, awb, abi, out);
  }
}
